// Model_74268574482845
// MI455X (gfx1250) — matrix operand order repaired and checked by reading, not compiled, not run on hardware
//
#include <hip/hip_runtime.h>

// ---------------------------------------------------------------------------
// Types
// ---------------------------------------------------------------------------
typedef __bf16 bf16_t;
typedef __attribute__((ext_vector_type(16))) __bf16 v16bf;
typedef __attribute__((ext_vector_type(8)))  __bf16 v8bf;
typedef __attribute__((ext_vector_type(8)))  float  v8f;

#define L_SEQ      512
#define D_MODEL    1024
#define H_HEADS    8
#define DK_HEAD    128
#define DI_FFN     2048
#define NL_LAYERS  2
#define T_WIN      25
#define W_LOC      24
#define NUM_LBL    7
#define ROWS       (L_SEQ * T_WIN)   // 12800

// ---------------------------------------------------------------------------
// f32 -> bf16 cast (weights / activations)
// ---------------------------------------------------------------------------
__global__ void cast_f32_bf16(const float* __restrict__ s, bf16_t* __restrict__ d, size_t n) {
    size_t i = (size_t)blockIdx.x * blockDim.x + threadIdx.x;
    size_t stride = (size_t)gridDim.x * blockDim.x;
    for (; i < n; i += stride) d[i] = (bf16_t)s[i];
}

// ---------------------------------------------------------------------------
// Gather rows of dialog_embeds by idx -> f32 + bf16 activation buffers
// ---------------------------------------------------------------------------
__global__ __launch_bounds__(256) void gather_kernel(
    const float* __restrict__ emb, const int* __restrict__ idx,
    float* __restrict__ Xf, bf16_t* __restrict__ Xb) {
    int row = blockIdx.x;                 // 0 .. ROWS-1
    int src = idx[row];
    const float* srcp = emb + (size_t)src * D_MODEL;
    size_t base = (size_t)row * D_MODEL;
    for (int c = threadIdx.x; c < D_MODEL; c += 256) {
        float v = srcp[c];
        Xf[base + c] = v;
        Xb[base + c] = (bf16_t)v;
    }
}

// ---------------------------------------------------------------------------
// bf16 WMMA GEMM:  C(MxN,f32) = A(MxK,bf16) * B(KxN,bf16) [+bias] [relu]
// Block tile 128x128, 8 waves, each wave: 32x64 via 2x4 WMMA 16x16x32 accs.
// Double-buffered LDS; A tile staged with GLOBAL_LOAD_ASYNC_TO_LDS_B128
// (ASYNCcnt), B tile transposed via packed dword LDS stores.
// M, N multiples of 128; K multiple of 32.
// ---------------------------------------------------------------------------
#define LDS_STR 48   // padded bf16 row stride (96B, 32B-aligned rows)

__global__ __launch_bounds__(256) void gemm_bf16_kernel(
    const bf16_t* __restrict__ A, int lda,
    const bf16_t* __restrict__ B, int ldb,
    const float* __restrict__ bias,
    float* __restrict__ Cf, int ldcf,
    bf16_t* __restrict__ Cb, int ldcb,
    int K, int relu)
{
    __shared__ __align__(32) bf16_t As[2][128][LDS_STR];
    __shared__ __align__(32) bf16_t Bs[2][128][LDS_STR];   // transposed: Bs[n][k]

    const int tid  = threadIdx.x;
    const int lane = tid & 31;
    const int wave = tid >> 5;
    const int wm   = wave & 3;            // 4 row-waves
    const int wn   = wave >> 2;           // 2 col-waves
    const int bm   = blockIdx.x * 128;
    const int bn   = blockIdx.y * 128;

    const v8f vzero = {0.f,0.f,0.f,0.f,0.f,0.f,0.f,0.f};
    v8f acc[2][4];
#pragma unroll
    for (int i = 0; i < 2; ++i)
#pragma unroll
        for (int j = 0; j < 4; ++j) acc[i][j] = vzero;

    // ISA 16-bit A-matrix 16x32 fragment addressing (wave32)
    const int ar  = lane & 15;            // row within 16
    const int ak  = (lane >> 4) << 3;     // K base: 0 or 8 (plus +16 second chunk)
    // B fragment (32x16, stored transposed in LDS)
    const int bnl = lane & 15;            // col within 16
    const int bk  = (lane >> 4) << 3;     // K base: 0 or 8 (plus +16 second chunk)

    // staging assignments
    const int sa_r = tid >> 1;            // A: 128 rows, 2 chunks of 16 per row
    const int sa_c = (tid & 1) << 4;
    const int sb_kp = tid >> 4;           // B: 16 K-pairs (2 rows each)
    const int sb_c  = (tid & 15) << 3;    //    8 N-columns per thread

    // stage tile (k0) into LDS buffer `buf`
    auto stage = [&](int k0, int buf) {
        // --- A tile: async Global->LDS, 32B per thread (2 x b128) ---
        unsigned lds_a = (unsigned)(size_t)&As[buf][sa_r][sa_c];
        unsigned long long ga =
            (unsigned long long)(size_t)(A + (size_t)(bm + sa_r) * lda + (k0 + sa_c));
        asm volatile(
            "global_load_async_to_lds_b128 %0, %1, off\n\t"
            "global_load_async_to_lds_b128 %0, %1, off offset:16"
            :: "v"(lds_a), "v"(ga) : "memory");
        // --- B tile: 2 K-rows x 8 N, transpose with packed-dword stores ---
        v8bf r0 = *(const v8bf*)(B + (size_t)(k0 + 2 * sb_kp)     * ldb + (bn + sb_c));
        v8bf r1 = *(const v8bf*)(B + (size_t)(k0 + 2 * sb_kp + 1) * ldb + (bn + sb_c));
#pragma unroll
        for (int j = 0; j < 8; ++j) {
            union { bf16_t h[2]; unsigned u; } pk;
            pk.h[0] = r0[j]; pk.h[1] = r1[j];
            *(unsigned*)&Bs[buf][sb_c + j][2 * sb_kp] = pk.u;
        }
    };

    int buf = 0;
    stage(0, 0);

    for (int k0 = 0; k0 < K; k0 += 32) {
        asm volatile("s_wait_asynccnt 0" ::: "memory");
        __syncthreads();                 // publish buffer `buf`
        if (k0 + 32 < K) stage(k0 + 32, buf ^ 1);   // overlap with compute

        v16bf afr[2];
#pragma unroll
        for (int i = 0; i < 2; ++i) {
            const bf16_t* p = &As[buf][wm * 32 + i * 16 + ar][ak];
            v8bf lo = *(const v8bf*)p;
            v8bf hi = *(const v8bf*)(p + 16);
            afr[i] = __builtin_shufflevector(lo, hi,
                        0,1,2,3,4,5,6,7,8,9,10,11,12,13,14,15);
        }
#pragma unroll
        for (int j = 0; j < 4; ++j) {
            const bf16_t* p = &Bs[buf][wn * 64 + j * 16 + bnl][bk];
            v8bf lo = *(const v8bf*)p;
            v8bf hi = *(const v8bf*)(p + 16);
            v16bf bfr = __builtin_shufflevector(lo, hi,
                        0,1,2,3,4,5,6,7,8,9,10,11,12,13,14,15);
#pragma unroll
            for (int i = 0; i < 2; ++i) {
                acc[i][j] = __builtin_amdgcn_wmma_f32_16x16x32_bf16(
                    false, afr[i], false, bfr, (short)0, acc[i][j], false, false);
            }
        }
        buf ^= 1;
    }

    // C layout: VGPR e -> row (lane<16 ? e : 8+e), col = lane&15
    const int rsel = (lane >> 4) << 3;
#pragma unroll
    for (int i = 0; i < 2; ++i) {
#pragma unroll
        for (int j = 0; j < 4; ++j) {
            const int col = bn + wn * 64 + j * 16 + (lane & 15);
            const float bv = bias ? bias[col] : 0.f;
#pragma unroll
            for (int e = 0; e < 8; ++e) {
                const int row = bm + wm * 32 + i * 16 + rsel + e;
                float v = acc[i][j][e] + bv;
                if (relu) v = fmaxf(v, 0.f);
                if (Cf) Cf[(size_t)row * ldcf + col] = v;
                if (Cb) Cb[(size_t)row * ldcb + col] = (bf16_t)v;
            }
        }
    }
}

// ---------------------------------------------------------------------------
// Masked softmax attention, per (b,h) block.  T=25, DK=128, wave32.
// ---------------------------------------------------------------------------
__global__ __launch_bounds__(256) void attn_kernel(
    const bf16_t* __restrict__ Qb, const bf16_t* __restrict__ Kb,
    const bf16_t* __restrict__ Vb, const unsigned char* __restrict__ mask,
    bf16_t* __restrict__ Ob)
{
    const int b = blockIdx.x / H_HEADS;
    const int h = blockIdx.x % H_HEADS;
    __shared__ float Ks[T_WIN][DK_HEAD];
    __shared__ float Vs[T_WIN][DK_HEAD];

    const int tid  = threadIdx.x;
    const int lane = tid & 31;
    const int warp = tid >> 5;

    for (int idx = tid; idx < T_WIN * DK_HEAD; idx += 256) {
        int j = idx >> 7, d = idx & 127;
        size_t off = ((size_t)(b * T_WIN + j)) * D_MODEL + h * DK_HEAD + d;
        Ks[j][d] = (float)Kb[off];
        Vs[j][d] = (float)Vb[off];
    }
    __syncthreads();

    const float scale = 0.088388347648318447f;   // 1/sqrt(128)

    for (int i0 = 0; i0 < T_WIN; i0 += 8) {
        int i = i0 + warp;
        if (i < T_WIN) {
            size_t qoff = ((size_t)(b * T_WIN + i)) * D_MODEL + h * DK_HEAD;
            float q[4];
#pragma unroll
            for (int c = 0; c < 4; ++c) q[c] = (float)Qb[qoff + lane + c * 32];

            float myS = -1e9f;
            for (int j = 0; j < T_WIN; ++j) {
                float p = 0.f;
#pragma unroll
                for (int c = 0; c < 4; ++c) p += q[c] * Ks[j][lane + c * 32];
#pragma unroll
                for (int o = 16; o > 0; o >>= 1) p += __shfl_down(p, o);
                p = __shfl(p, 0) * scale;
                float sc = (mask[b * T_WIN + j] != 0) ? p : -1e9f;
                if (lane == j) myS = sc;
            }
            // warp softmax over 25 values (lanes 0..24)
            float s = (lane < T_WIN) ? myS : -3.0e38f;
            float mx = s;
#pragma unroll
            for (int o = 16; o > 0; o >>= 1) mx = fmaxf(mx, __shfl_xor(mx, o));
            float e = (lane < T_WIN) ? __expf(s - mx) : 0.f;
            float sum = e;
#pragma unroll
            for (int o = 16; o > 0; o >>= 1) sum += __shfl_xor(sum, o);
            float a = e / sum;

            float o4[4] = {0.f, 0.f, 0.f, 0.f};
            for (int j = 0; j < T_WIN; ++j) {
                float aj = __shfl(a, j);
#pragma unroll
                for (int c = 0; c < 4; ++c) o4[c] += aj * Vs[j][lane + c * 32];
            }
#pragma unroll
            for (int c = 0; c < 4; ++c)
                Ob[qoff + lane + c * 32] = (bf16_t)o4[c];
        }
    }
}

// ---------------------------------------------------------------------------
// Fused residual add + LayerNorm. X (f32) updated in place, Xb bf16 copy.
// ---------------------------------------------------------------------------
__device__ __forceinline__ float block_reduce_sum256(float v, float* red) {
    int lane = threadIdx.x & 31, warp = threadIdx.x >> 5;
#pragma unroll
    for (int o = 16; o > 0; o >>= 1) v += __shfl_down(v, o);
    if (lane == 0) red[warp] = v;
    __syncthreads();
    float s = 0.f;
    if (warp == 0) {
        s = (lane < 8) ? red[lane] : 0.f;
#pragma unroll
        for (int o = 4; o > 0; o >>= 1) s += __shfl_down(s, o);
        if (lane == 0) red[0] = s;
    }
    __syncthreads();
    s = red[0];
    __syncthreads();
    return s;
}

__global__ __launch_bounds__(256) void ln_add_kernel(
    const float* __restrict__ O, float* __restrict__ X, bf16_t* __restrict__ Xb,
    const float* __restrict__ g, const float* __restrict__ b)
{
    __shared__ float red[8];
    size_t row = blockIdx.x;
    int tid = threadIdx.x;
    float v[4];
    float s = 0.f;
#pragma unroll
    for (int c = 0; c < 4; ++c) {
        int col = tid + c * 256;
        v[c] = O[row * D_MODEL + col] + X[row * D_MODEL + col];
        s += v[c];
    }
    float mean = block_reduce_sum256(s, red) * (1.f / D_MODEL);
    float s2 = 0.f;
#pragma unroll
    for (int c = 0; c < 4; ++c) { float d = v[c] - mean; s2 += d * d; }
    float var = block_reduce_sum256(s2, red) * (1.f / D_MODEL);
    float inv = rsqrtf(var + 1e-6f);
#pragma unroll
    for (int c = 0; c < 4; ++c) {
        int col = tid + c * 256;
        float y = (v[c] - mean) * inv * g[col] + b[col];
        X[row * D_MODEL + col]  = y;
        Xb[row * D_MODEL + col] = (bf16_t)y;
    }
}

// ---------------------------------------------------------------------------
// Attention pooling per dialog. Writes bf16 halves of concat buffers.
// ---------------------------------------------------------------------------
__global__ __launch_bounds__(256) void pool_kernel(
    const float* __restrict__ X,      // enc f32 (ROWS x D)
    const float* __restrict__ proj,   // (ROWS x D) f32, rows w<24 used
    const unsigned char* __restrict__ mask,
    bf16_t* __restrict__ outcat_b,    // (L x 2D)
    bf16_t* __restrict__ statecat_b,  // (L x 2D)
    int off)
{
    int l = blockIdx.x;
    __shared__ float outv[D_MODEL];
    __shared__ float sraw[W_LOC];
    __shared__ float wnorm[W_LOC];
    int tid = threadIdx.x, lane = tid & 31, warp = tid >> 5;

    const float* outrow = X + ((size_t)l * T_WIN + W_LOC) * D_MODEL;
    for (int c = tid; c < D_MODEL; c += 256) {
        float v = outrow[c];
        outv[c] = v;
        outcat_b[(size_t)l * (2 * D_MODEL) + off + c] = (bf16_t)v;
    }
    __syncthreads();

    for (int w = warp; w < W_LOC; w += 8) {
        const float* pr = proj + ((size_t)l * T_WIN + w) * D_MODEL;
        float p = 0.f;
        for (int d = lane; d < D_MODEL; d += 32) p += pr[d] * outv[d];
#pragma unroll
        for (int o = 16; o > 0; o >>= 1) p += __shfl_down(p, o);
        if (lane == 0) {
            float sc = tanhf(p);
            sraw[w] = (mask[l * T_WIN + w] != 0) ? sc : -1e9f;
        }
    }
    __syncthreads();

    if (warp == 0) {
        float s = (lane < W_LOC) ? sraw[lane] : -3.0e38f;
        float mx = s;
#pragma unroll
        for (int o = 16; o > 0; o >>= 1) mx = fmaxf(mx, __shfl_xor(mx, o));
        float e = (lane < W_LOC) ? __expf(s - mx) : 0.f;
        float sum = e;
#pragma unroll
        for (int o = 16; o > 0; o >>= 1) sum += __shfl_xor(sum, o);
        float a = e / sum;
        if (lane < W_LOC)
            wnorm[lane] = (mask[l * T_WIN + lane] != 0) ? a : 0.f;  // softmax * lm
    }
    __syncthreads();

    for (int c = tid; c < D_MODEL; c += 256) {
        float st = 0.f;
#pragma unroll 4
        for (int w = 0; w < W_LOC; ++w)
            st += wnorm[w] * X[((size_t)l * T_WIN + w) * D_MODEL + c];
        statecat_b[(size_t)l * (2 * D_MODEL) + off + c] = (bf16_t)st;
    }
}

// ---------------------------------------------------------------------------
// Sequential dual-GRU scan (512 steps). blockIdx.x: 0=ctx GRU, 1=spk GRU.
// ---------------------------------------------------------------------------
__global__ __launch_bounds__(1024) void gru_kernel(
    const float* __restrict__ x3_c, const float* __restrict__ x3_s,
    const float* __restrict__ Wc, const float* __restrict__ bc,
    const float* __restrict__ Ws, const float* __restrict__ bs,
    bf16_t* __restrict__ hxcat_b)     // (L x 2D)
{
    const int e = blockIdx.x;
    const float* x3   = e ? x3_s : x3_c;
    const float* W    = e ? Ws : Wc;
    const float* bias = e ? bs : bc;
    __shared__ float h[D_MODEL];
    const int d = threadIdx.x;
    h[d] = 0.f;
    __syncthreads();

    for (int t = 0; t < L_SEQ; ++t) {
        float a0 = 0.f, a1 = 0.f, a2 = 0.f;
        for (int k = 0; k < D_MODEL; ++k) {
            float hk = h[k];
            const float* wr = W + (size_t)k * (3 * D_MODEL);
            a0 += hk * wr[d];
            a1 += hk * wr[d + D_MODEL];
            a2 += hk * wr[d + 2 * D_MODEL];
        }
        const float* xt = x3 + (size_t)t * (3 * D_MODEL);
        float r = 1.f / (1.f + __expf(-(xt[d] + a0 + bias[d])));
        float u = 1.f / (1.f + __expf(-(xt[d + D_MODEL] + a1 + bias[d + D_MODEL])));
        float n = tanhf(xt[d + 2 * D_MODEL] + r * (a2 + bias[d + 2 * D_MODEL]));
        float hnew = u * h[d] + (1.f - u) * n;
        __syncthreads();
        h[d] = hnew;
        hxcat_b[(size_t)t * (2 * D_MODEL) + e * D_MODEL + d] = (bf16_t)hnew;
        __syncthreads();
    }
}

// ---------------------------------------------------------------------------
// Final classifier: (L x 3072) @ (3072 x 7) + bias
// ---------------------------------------------------------------------------
__global__ __launch_bounds__(256) void fc_kernel(
    const float* __restrict__ F, const float* __restrict__ Wfc,
    const float* __restrict__ bfc, float* __restrict__ out)
{
    __shared__ float red[8];
    int l = blockIdx.x;
    int tid = threadIdx.x, lane = tid & 31, warp = tid >> 5;
    for (int j = 0; j < NUM_LBL; ++j) {
        float p = 0.f;
        for (int i = tid; i < 3 * D_MODEL; i += 256)
            p += F[(size_t)l * (3 * D_MODEL) + i] * Wfc[(size_t)i * NUM_LBL + j];
#pragma unroll
        for (int o = 16; o > 0; o >>= 1) p += __shfl_down(p, o);
        if (lane == 0) red[warp] = p;
        __syncthreads();
        if (tid == 0) {
            float s = 0.f;
#pragma unroll
            for (int w = 0; w < 8; ++w) s += red[w];
            out[(size_t)l * NUM_LBL + j] = s + bfc[j];
        }
        __syncthreads();
    }
}

// ---------------------------------------------------------------------------
// Launch orchestration
// ---------------------------------------------------------------------------
extern "C" void kernel_launch(void* const* d_in, const int* in_sizes, int n_in,
                              void* d_out, int out_size, void* d_ws, size_t ws_size,
                              hipStream_t stream) {
    (void)in_sizes; (void)n_in; (void)out_size; (void)ws_size;

    const float* emb      = (const float*)d_in[0];
    const int*   ctx_idx  = (const int*)d_in[1];
    const unsigned char* ctx_mask = (const unsigned char*)d_in[2];
    const int*   spk_idx  = (const int*)d_in[3];
    const unsigned char* spk_mask = (const unsigned char*)d_in[4];

    const float* p_Wq   = (const float*)d_in[5];
    const float* p_Wk   = (const float*)d_in[6];
    const float* p_Wv   = (const float*)d_in[7];
    const float* p_Wo   = (const float*)d_in[8];
    const float* p_ln1g = (const float*)d_in[9];
    const float* p_ln1b = (const float*)d_in[10];
    const float* p_W1e  = (const float*)d_in[11];
    const float* p_b1e  = (const float*)d_in[12];
    const float* p_W2e  = (const float*)d_in[13];
    const float* p_b2e  = (const float*)d_in[14];
    const float* p_ln2g = (const float*)d_in[15];
    const float* p_ln2b = (const float*)d_in[16];
    const float* p_W1w  = (const float*)d_in[17];
    const float* p_W1b  = (const float*)d_in[18];
    const float* p_W2w  = (const float*)d_in[19];
    const float* p_W2b  = (const float*)d_in[20];
    const float* p_W3w  = (const float*)d_in[21];
    const float* p_W3b  = (const float*)d_in[22];
    const float* p_W4w  = (const float*)d_in[23];
    const float* p_W4b  = (const float*)d_in[24];
    const float* p_x2hc_w = (const float*)d_in[25];
    const float* p_x2hc_b = (const float*)d_in[26];
    const float* p_h2hc_w = (const float*)d_in[27];
    const float* p_h2hc_b = (const float*)d_in[28];
    const float* p_x2hs_w = (const float*)d_in[29];
    const float* p_x2hs_b = (const float*)d_in[30];
    const float* p_h2hs_w = (const float*)d_in[31];
    const float* p_h2hs_b = (const float*)d_in[32];
    const float* p_fcw  = (const float*)d_in[33];
    const float* p_fcb  = (const float*)d_in[34];
    float* out = (float*)d_out;

    // ---- bump allocator over workspace ----
    char* ws = (char*)d_ws;
    size_t off = 0;
    auto alloc = [&](size_t bytes) -> void* {
        void* p = ws + off;
        off = (off + bytes + 255) & ~(size_t)255;
        return p;
    };

    const size_t DD   = (size_t)D_MODEL * D_MODEL;      // 1M
    const size_t DDI  = (size_t)D_MODEL * DI_FFN;       // 2M
    const size_t RD   = (size_t)ROWS * D_MODEL;         // 13.1M
    const size_t RDI  = (size_t)ROWS * DI_FFN;

    // bf16 weights
    bf16_t* Wq_b  = (bf16_t*)alloc(NL_LAYERS * DD  * 2);
    bf16_t* Wk_b  = (bf16_t*)alloc(NL_LAYERS * DD  * 2);
    bf16_t* Wv_b  = (bf16_t*)alloc(NL_LAYERS * DD  * 2);
    bf16_t* Wo_b  = (bf16_t*)alloc(NL_LAYERS * DD  * 2);
    bf16_t* W1e_b = (bf16_t*)alloc(NL_LAYERS * DDI * 2);
    bf16_t* W2e_b = (bf16_t*)alloc(NL_LAYERS * DDI * 2);
    bf16_t* W1w_b = (bf16_t*)alloc(DD * 2);
    bf16_t* W2w_b = (bf16_t*)alloc(2 * DD * 2);
    bf16_t* W3w_b = (bf16_t*)alloc(2 * DD * 2);
    bf16_t* W4w_b = (bf16_t*)alloc(2 * DD * 2);
    bf16_t* x2hc_b = (bf16_t*)alloc(3 * DD * 2);
    bf16_t* x2hs_b = (bf16_t*)alloc(3 * DD * 2);

    // per-encoder persistent activations
    float*  Xf[2]; bf16_t* Xb[2];
    Xf[0] = (float*)alloc(RD * 4);  Xb[0] = (bf16_t*)alloc(RD * 2);
    Xf[1] = (float*)alloc(RD * 4);  Xb[1] = (bf16_t*)alloc(RD * 2);

    // shared scratch
    bf16_t* Qb = (bf16_t*)alloc(RD * 2);
    bf16_t* Kb = (bf16_t*)alloc(RD * 2);
    bf16_t* Vb = (bf16_t*)alloc(RD * 2);
    bf16_t* Ob = (bf16_t*)alloc(RD * 2);
    float*  Of = (float*)alloc(RD * 4);
    bf16_t* Hb = (bf16_t*)alloc(RDI * 2);
    float*  projf = (float*)alloc(RD * 4);

    bf16_t* outcat_b   = (bf16_t*)alloc((size_t)L_SEQ * 2 * D_MODEL * 2);
    bf16_t* statecat_b = (bf16_t*)alloc((size_t)L_SEQ * 2 * D_MODEL * 2);
    bf16_t* hxcat_b    = (bf16_t*)alloc((size_t)L_SEQ * 2 * D_MODEL * 2);
    float*  x3c = (float*)alloc((size_t)L_SEQ * 3 * D_MODEL * 4);
    float*  x3s = (float*)alloc((size_t)L_SEQ * 3 * D_MODEL * 4);
    float*  finalf = (float*)alloc((size_t)L_SEQ * 3 * D_MODEL * 4);

    auto cast = [&](const float* s, bf16_t* d, size_t n) {
        cast_f32_bf16<<<512, 256, 0, stream>>>(s, d, n);
    };
    auto gemm = [&](const bf16_t* A, int lda, const bf16_t* B, int ldb,
                    const float* bias, float* Cf, int ldcf, bf16_t* Cb, int ldcb,
                    int M, int N, int K, int relu) {
        dim3 g(M / 128, N / 128);
        gemm_bf16_kernel<<<g, dim3(256), 0, stream>>>(A, lda, B, ldb, bias,
                                                      Cf, ldcf, Cb, ldcb, K, relu);
    };

    // ---- weight casts ----
    cast(p_Wq,  Wq_b,  NL_LAYERS * DD);
    cast(p_Wk,  Wk_b,  NL_LAYERS * DD);
    cast(p_Wv,  Wv_b,  NL_LAYERS * DD);
    cast(p_Wo,  Wo_b,  NL_LAYERS * DD);
    cast(p_W1e, W1e_b, NL_LAYERS * DDI);
    cast(p_W2e, W2e_b, NL_LAYERS * DDI);
    cast(p_W1w, W1w_b, DD);
    cast(p_W2w, W2w_b, 2 * DD);
    cast(p_W3w, W3w_b, 2 * DD);
    cast(p_W4w, W4w_b, 2 * DD);
    cast(p_x2hc_w, x2hc_b, 3 * DD);
    cast(p_x2hs_w, x2hs_b, 3 * DD);

    const int* idxs[2]  = {ctx_idx, spk_idx};
    const unsigned char* masks[2] = {ctx_mask, spk_mask};

    // ---- encoders ----
    for (int e = 0; e < 2; ++e) {
        gather_kernel<<<ROWS, 256, 0, stream>>>(emb, idxs[e], Xf[e], Xb[e]);
        for (int l = 0; l < NL_LAYERS; ++l) {
            const bf16_t* wq = Wq_b + (size_t)l * DD;
            const bf16_t* wk = Wk_b + (size_t)l * DD;
            const bf16_t* wv = Wv_b + (size_t)l * DD;
            const bf16_t* wo = Wo_b + (size_t)l * DD;
            gemm(Xb[e], D_MODEL, wq, D_MODEL, nullptr, nullptr, 0, Qb, D_MODEL,
                 ROWS, D_MODEL, D_MODEL, 0);
            gemm(Xb[e], D_MODEL, wk, D_MODEL, nullptr, nullptr, 0, Kb, D_MODEL,
                 ROWS, D_MODEL, D_MODEL, 0);
            gemm(Xb[e], D_MODEL, wv, D_MODEL, nullptr, nullptr, 0, Vb, D_MODEL,
                 ROWS, D_MODEL, D_MODEL, 0);
            attn_kernel<<<L_SEQ * H_HEADS, 256, 0, stream>>>(Qb, Kb, Vb, masks[e], Ob);
            gemm(Ob, D_MODEL, wo, D_MODEL, nullptr, Of, D_MODEL, nullptr, 0,
                 ROWS, D_MODEL, D_MODEL, 0);
            ln_add_kernel<<<ROWS, 256, 0, stream>>>(Of, Xf[e], Xb[e],
                p_ln1g + (size_t)l * D_MODEL, p_ln1b + (size_t)l * D_MODEL);
            // FFN
            gemm(Xb[e], D_MODEL, W1e_b + (size_t)l * DDI, DI_FFN,
                 p_b1e + (size_t)l * DI_FFN, nullptr, 0, Hb, DI_FFN,
                 ROWS, DI_FFN, D_MODEL, 1);
            gemm(Hb, DI_FFN, W2e_b + (size_t)l * DDI, D_MODEL,
                 p_b2e + (size_t)l * D_MODEL, Of, D_MODEL, nullptr, 0,
                 ROWS, D_MODEL, DI_FFN, 0);
            ln_add_kernel<<<ROWS, 256, 0, stream>>>(Of, Xf[e], Xb[e],
                p_ln2g + (size_t)l * D_MODEL, p_ln2b + (size_t)l * D_MODEL);
        }
    }

    // ---- pooling ----
    for (int e = 0; e < 2; ++e) {
        gemm(Xb[e], D_MODEL, W1w_b, D_MODEL, p_W1b, projf, D_MODEL, nullptr, 0,
             ROWS, D_MODEL, D_MODEL, 0);
        pool_kernel<<<L_SEQ, 256, 0, stream>>>(Xf[e], projf, masks[e],
                                               outcat_b, statecat_b, e * D_MODEL);
    }

    // ---- x2h projections ----
    gemm(statecat_b, 2 * D_MODEL, x2hc_b, 3 * D_MODEL, p_x2hc_b,
         x3c, 3 * D_MODEL, nullptr, 0, L_SEQ, 3 * D_MODEL, D_MODEL, 0);
    gemm(statecat_b + D_MODEL, 2 * D_MODEL, x2hs_b, 3 * D_MODEL, p_x2hs_b,
         x3s, 3 * D_MODEL, nullptr, 0, L_SEQ, 3 * D_MODEL, D_MODEL, 0);

    // ---- sequential GRU scan ----
    gru_kernel<<<2, 1024, 0, stream>>>(x3c, x3s, p_h2hc_w, p_h2hc_b,
                                       p_h2hs_w, p_h2hs_b, hxcat_b);

    // ---- head projections into finalf (L x 3072) ----
    gemm(outcat_b,   2 * D_MODEL, W2w_b, D_MODEL, p_W2b,
         finalf + 0,            3 * D_MODEL, nullptr, 0, L_SEQ, D_MODEL, 2 * D_MODEL, 0);
    gemm(statecat_b, 2 * D_MODEL, W3w_b, D_MODEL, p_W3b,
         finalf + D_MODEL,      3 * D_MODEL, nullptr, 0, L_SEQ, D_MODEL, 2 * D_MODEL, 0);
    gemm(hxcat_b,    2 * D_MODEL, W4w_b, D_MODEL, p_W4b,
         finalf + 2 * D_MODEL,  3 * D_MODEL, nullptr, 0, L_SEQ, D_MODEL, 2 * D_MODEL, 0);

    // ---- classifier ----
    fc_kernel<<<L_SEQ, 256, 0, stream>>>(finalf, p_fcw, p_fcb, out);
}
